// BahdanauAttention_29454885716301
// MI455X (gfx1250) — matrix operand order repaired and checked by reading, not compiled, not run on hardware
//
#include <hip/hip_runtime.h>
#include <hip/hip_bf16.h>
#include <math.h>

// ---------------------------------------------------------------------------
// Types for CDNA5 WMMA (wave32): D(f32 16x16) = A(bf16 16x32) x B(bf16 32x16) + C
// ---------------------------------------------------------------------------
typedef __bf16 v16bf __attribute__((ext_vector_type(16)));
typedef float  v8f   __attribute__((ext_vector_type(8)));

union Frag16 {          // 32 bytes: 8 VGPRs holding 16 bf16 values
    uint4  q[2];
    v16bf  v;
};

__device__ __forceinline__ unsigned short f32_to_bf16(float f) {
    unsigned int u = __float_as_uint(f);
    unsigned int r = u + 0x7FFFu + ((u >> 16) & 1u);   // round-to-nearest-even
    return (unsigned short)(r >> 16);
}

// Problem sizes (fixed by the reference)
#define BB     32
#define TT     2048
#define DQ     1024
#define DV     1024
#define UNITS  1024

// ---------------------------------------------------------------------------
// Kernel 0: Ws [DV,UNITS] f32 (row-major, k-major) -> WsP bf16 [u][k]
// (transposed so each WMMA B-fragment is 2 contiguous b128 loads per lane)
// ---------------------------------------------------------------------------
__global__ __launch_bounds__(256) void packWs_kernel(const float* __restrict__ Ws,
                                                     unsigned short* __restrict__ wsP) {
    __shared__ float tile[32][33];
    const int k0 = blockIdx.x * 32;
    const int u0 = blockIdx.y * 32;
    const int tx = threadIdx.x & 31;
    const int ty = threadIdx.x >> 5;          // 0..7
#pragma unroll
    for (int i = 0; i < 4; ++i) {
        int kr = ty * 4 + i;
        tile[kr][tx] = Ws[(size_t)(k0 + kr) * UNITS + (u0 + tx)];
    }
    __syncthreads();
#pragma unroll
    for (int i = 0; i < 4; ++i) {
        int ur = ty * 4 + i;
        // wsP[u][k] = Ws[k][u]
        wsP[(size_t)(u0 + ur) * DV + (k0 + tx)] = f32_to_bf16(tile[tx][ur]);
    }
}

// ---------------------------------------------------------------------------
// Kernel 1: qsum[b][u] = query[b]@Wh[:,u] + bh[u] + bs[u] + bc[u]   (fp32)
// ---------------------------------------------------------------------------
__global__ __launch_bounds__(256) void qsum_kernel(const float* __restrict__ query,
                                                   const float* __restrict__ Wh,
                                                   const float* __restrict__ bh,
                                                   const float* __restrict__ bs,
                                                   const float* __restrict__ bc,
                                                   float* __restrict__ qsum) {
    const int idx = blockIdx.x * 256 + threadIdx.x;    // 32768 outputs
    const int b = idx >> 10;
    const int u = idx & 1023;
    const float* __restrict__ q = query + (size_t)b * DQ;
    float acc = bh[u] + bs[u] + bc[u];
#pragma unroll 4
    for (int k = 0; k < DQ; ++k)
        acc += q[k] * Wh[(size_t)k * UNITS + u];
    qsum[idx] = acc;
}

// ---------------------------------------------------------------------------
// Kernel 2 (the WMMA kernel):
//   scores[b][t] = sum_u tanh( values[b,t,:]@Ws[:,u] + qsum[b,u] + cov[b,t]*Wc[u] ) * Vw[u] + Vb
// 1 WG = 32 rows of t (one b), 128 threads = 4 waves:
//   wave = (row-tile rt in {0,1}) x (u-half uh in {0,1}; 32 u-tiles each)
// values staged in LDS pre-swizzled to WMMA A-fragment layout (16-bit A 16x32).
// ---------------------------------------------------------------------------
__global__ __launch_bounds__(128) void score_kernel(const float* __restrict__ values,
                                                    const float* __restrict__ prevcov,
                                                    const float* __restrict__ qsum,
                                                    const float* __restrict__ Wc,
                                                    const float* __restrict__ Vw,
                                                    const float* __restrict__ Vb,
                                                    const unsigned short* __restrict__ wsP,
                                                    float* __restrict__ scores) {
    __shared__ unsigned int ldsA[16384];     // 64 KB: 32 rows x 1024 K, bf16, fragment order
    const int wg  = blockIdx.x;              // 0..2047
    const int b   = wg >> 6;                 // 64 blocks per batch
    const int t0  = (wg & 63) * 32;
    const int tid = threadIdx.x;

    // ---- stage values[b, t0:t0+32, :] -> bf16 in A-fragment order ----------
    // A-frag (16-bit 16x32): lanes 0-15: M=lane, VGPR0-3 K=0..7, VGPR4-7 K=16..23
    //                        lanes16-31: M=lane-16, VGPR0-3 K=8..15, VGPR4-7 K=24..31
    {
        const float2* __restrict__ src =
            (const float2*)(values + ((size_t)b * TT + t0) * DV);
        for (int p = tid; p < 16384; p += 128) {     // 16384 k-pairs
            const int mloc = p >> 9;                 // local row 0..31
            const int kp   = p & 511;
            const float2 v = src[(size_t)mloc * 512 + kp];
            const unsigned int pk =
                (unsigned int)f32_to_bf16(v.x) | ((unsigned int)f32_to_bf16(v.y) << 16);
            const int k  = kp << 1;
            const int rt = mloc >> 4, m = mloc & 15;
            const int kc = k >> 5,   kk = k & 31;
            const int h  = (kk >> 3) & 1;                       // lane half
            const int vg = ((kk >= 16) ? 4 : 0) + ((kk & 7) >> 1); // VGPR 0..7
            const int lane = h * 16 + m;
            // chunk = 256 dwords; group(VGPR0-3 / 4-7) = 128 dwords; lane*16B
            const int idx = rt * 8192 + kc * 256 + (vg >> 2) * 128 + lane * 4 + (vg & 3);
            ldsA[idx] = pk;
        }
    }
    __syncthreads();

    const int wv   = tid >> 5;       // wave 0..3
    const int lane = tid & 31;
    const int rt   = wv >> 1;        // row-tile 0/1 (16 t-rows each)
    const int uh   = wv & 1;         // u-half: u-tiles [uh*32, uh*32+32)
    const int n    = lane & 15;      // C/B column within tile
    const int half = lane >> 4;      // C rows r / r+8

    // coverage for this lane's 8 C rows: t = t0 + rt*16 + half*8 + r
    float cov[8];
    {
        const float* __restrict__ pc = prevcov + (size_t)b * TT + t0 + rt * 16 + half * 8;
#pragma unroll
        for (int r = 0; r < 8; ++r) cov[r] = pc[r];
    }

    float scorep[8] = {0.f, 0.f, 0.f, 0.f, 0.f, 0.f, 0.f, 0.f};
    const int aBase = rt * 8192;

    for (int g = 0; g < 4; ++g) {                 // 4 groups of 8 u-tiles
        const int ut0 = uh * 32 + g * 8;
        v8f acc[8];
#pragma unroll
        for (int j = 0; j < 8; ++j) acc[j] = (v8f){0.f, 0.f, 0.f, 0.f, 0.f, 0.f, 0.f, 0.f};

        for (int kc = 0; kc < 32; ++kc) {         // K = 1024 in chunks of 32
            Frag16 af;
            af.q[0] = *(const uint4*)&ldsA[aBase + kc * 256 + lane * 4];
            af.q[1] = *(const uint4*)&ldsA[aBase + kc * 256 + 128 + lane * 4];
#pragma unroll
            for (int j = 0; j < 8; ++j) {
                // B-frag (bf16 32x16): lanes 0-15 -> N=lane, K kc*32+0..7, 16..23;
                //                      lanes16-31 -> N=lane-16, K kc*32+8..15, 24..31
                const int u = (ut0 + j) * 16 + n;
                const uint4* __restrict__ bp =
                    (const uint4*)(wsP + (size_t)u * DV + kc * 32 + half * 8);
                Frag16 bfr;
                bfr.q[0] = bp[0];
                bfr.q[1] = bp[2];
                acc[j] = __builtin_amdgcn_wmma_f32_16x16x32_bf16(
                    false, af.v, false, bfr.v, (short)0, acc[j], false, false);
            }
        }
        // epilogue: tanh MLP reduction over this group's 128 u columns
#pragma unroll
        for (int j = 0; j < 8; ++j) {
            const int u   = (ut0 + j) * 16 + n;
            const float qv  = qsum[(size_t)b * UNITS + u];
            const float wcv = Wc[u];
            const float vwv = Vw[u];
#pragma unroll
            for (int r = 0; r < 8; ++r) {
                const float pre = acc[j][r] + qv + cov[r] * wcv;
                scorep[r] += tanhf(pre) * vwv;
            }
        }
    }

    // ---- reduce per-lane partial scores -> 32 row scores -------------------
    __syncthreads();                       // all waves done reading ldsA
    float* ps = (float*)ldsA;              // reuse: ps[uh*32 + local_row]
#pragma unroll
    for (int r = 0; r < 8; ++r) {
        float s = scorep[r];
        s += __shfl_xor(s, 1, 32);
        s += __shfl_xor(s, 2, 32);
        s += __shfl_xor(s, 4, 32);
        s += __shfl_xor(s, 8, 32);         // sum over the 16 lanes of this half
        if (n == 0) ps[uh * 32 + rt * 16 + half * 8 + r] = s;
    }
    __syncthreads();
    if (tid < 32) {
        const float s = ps[tid] + ps[32 + tid] + Vb[0];
        scores[(size_t)b * TT + t0 + tid] = s;
    }
}

// ---------------------------------------------------------------------------
// Kernel 3: softmax over T per batch; write attention + coverage outputs
// ---------------------------------------------------------------------------
__global__ __launch_bounds__(256) void softmax_kernel(const float* __restrict__ scores,
                                                      const float* __restrict__ prevcov,
                                                      float* __restrict__ attn,
                                                      float* __restrict__ coverage) {
    __shared__ float red[256];
    const int b = blockIdx.x, tid = threadIdx.x;
    const float* __restrict__ s = scores + (size_t)b * TT;
    float loc[8];
    float m = -3.4e38f;
#pragma unroll
    for (int i = 0; i < 8; ++i) {
        loc[i] = s[tid + i * 256];
        m = fmaxf(m, loc[i]);
    }
    red[tid] = m;
    __syncthreads();
    for (int st = 128; st > 0; st >>= 1) {
        if (tid < st) red[tid] = fmaxf(red[tid], red[tid + st]);
        __syncthreads();
    }
    m = red[0];
    __syncthreads();
    float sum = 0.f;
#pragma unroll
    for (int i = 0; i < 8; ++i) {
        loc[i] = expf(loc[i] - m);
        sum += loc[i];
    }
    red[tid] = sum;
    __syncthreads();
    for (int st = 128; st > 0; st >>= 1) {
        if (tid < st) red[tid] += red[tid + st];
        __syncthreads();
    }
    const float inv = 1.0f / red[0];
#pragma unroll
    for (int i = 0; i < 8; ++i) {
        const int t = tid + i * 256;
        const float a = loc[i] * inv;
        attn[(size_t)b * TT + t] = a;
        coverage[(size_t)b * TT + t] = a + prevcov[(size_t)b * TT + t];
    }
}

// ---------------------------------------------------------------------------
// Kernel 4: context[b][d] = sum_t attn[b][t] * values[b][t][d]   (mem-bound)
// ---------------------------------------------------------------------------
__global__ __launch_bounds__(256) void context_kernel(const float* __restrict__ values,
                                                      const float* __restrict__ attn,
                                                      float* __restrict__ ctx) {
    __shared__ float aw[TT];
    const int b = blockIdx.y;
    const int d = blockIdx.x * 256 + threadIdx.x;
    for (int i = threadIdx.x; i < TT; i += 256) aw[i] = attn[(size_t)b * TT + i];
    __syncthreads();
    const float* __restrict__ vp = values + (size_t)b * TT * DV + d;
    float acc = 0.f;
    for (int t = 0; t < TT; t += 4) {
        acc += aw[t + 0] * vp[(size_t)(t + 0) * DV];
        acc += aw[t + 1] * vp[(size_t)(t + 1) * DV];
        acc += aw[t + 2] * vp[(size_t)(t + 2) * DV];
        acc += aw[t + 3] * vp[(size_t)(t + 3) * DV];
    }
    ctx[(size_t)b * DV + d] = acc;
}

// ---------------------------------------------------------------------------
// Launch. Inputs: 0 query, 1 values, 2 prev_coverage, 3 Wh, 4 bh, 5 Ws, 6 bs,
//                 7 Wc, 8 bc, 9 Vw, 10 Vb.
// Output: [context 32x1024 | attn 32x2048 | coverage 32x2048] fp32.
// ---------------------------------------------------------------------------
extern "C" void kernel_launch(void* const* d_in, const int* in_sizes, int n_in,
                              void* d_out, int out_size, void* d_ws, size_t ws_size,
                              hipStream_t stream) {
    const float* query   = (const float*)d_in[0];
    const float* values  = (const float*)d_in[1];
    const float* prevcov = (const float*)d_in[2];
    const float* Wh      = (const float*)d_in[3];
    const float* bh      = (const float*)d_in[4];
    const float* Ws      = (const float*)d_in[5];
    const float* bs      = (const float*)d_in[6];
    const float* Wc      = (const float*)d_in[7];
    const float* bc      = (const float*)d_in[8];
    const float* Vw      = (const float*)d_in[9];
    const float* Vb      = (const float*)d_in[10];

    float* ctx  = (float*)d_out;                       // 32*1024
    float* attn = (float*)d_out + BB * DV;             // 32*2048
    float* cov  = (float*)d_out + BB * DV + BB * TT;   // 32*2048

    // workspace: WsP bf16 (2 MB) | qsum (128 KB) | scores (256 KB)
    unsigned short* wsP = (unsigned short*)d_ws;
    float* qsum   = (float*)((char*)d_ws + (2u << 20));
    float* scores = (float*)((char*)d_ws + (2u << 20) + (128u << 10));

    packWs_kernel<<<dim3(DV / 32, UNITS / 32), 256, 0, stream>>>(Ws, wsP);
    qsum_kernel<<<(BB * UNITS) / 256, 256, 0, stream>>>(query, Wh, bh, bs, bc, qsum);
    score_kernel<<<(BB * TT) / 32, 128, 0, stream>>>(values, prevcov, qsum, Wc, Vw, Vb,
                                                     wsP, scores);
    softmax_kernel<<<BB, 256, 0, stream>>>(scores, prevcov, attn, cov);
    context_kernel<<<dim3(DV / 256, BB), 256, 0, stream>>>(values, attn, ctx);
}
